// SubGI_5944234737772
// MI455X (gfx1250) — compile-verified, matrix operand order checked by reading, not run on hardware
//
#include <hip/hip_runtime.h>
#include <hip/hip_bf16.h>
#include <math.h>

typedef __attribute__((ext_vector_type(16))) __bf16       v16bf;
typedef __attribute__((ext_vector_type(8)))  float        v8f;
typedef __attribute__((ext_vector_type(8)))  unsigned int v8u;

__device__ inline unsigned pkbf(float a, float b) {
    union { __bf16 h[2]; unsigned u; } u;
    u.h[0] = (__bf16)a; u.h[1] = (__bf16)b;
    return u.u;
}

__device__ inline float softplus_f(float x) {
    return (x > 0.f) ? (x + log1pf(expf(-x))) : log1pf(expf(x));
}

// ---------------- utility kernels ----------------
__global__ void k_zero(float* __restrict__ p, long n) {
    long i = (long)blockIdx.x * blockDim.x + threadIdx.x;
    long st = (long)gridDim.x * blockDim.x;
    for (; i < n; i += st) p[i] = 0.f;
}

// segment_sum over edges: one wave per edge, 4 floats per lane (128 features)
__global__ void k_scatter_add(const float* __restrict__ src, const int* __restrict__ es,
                              const int* __restrict__ ed, float* __restrict__ out, int E) {
    int wid  = (blockIdx.x * blockDim.x + threadIdx.x) >> 5;
    int lane = threadIdx.x & 31;
    int nw   = (gridDim.x * blockDim.x) >> 5;
    for (int e = wid; e < E; e += nw) {
        int s = es[e], d = ed[e];
        const float4 v = *(const float4*)(src + (long)s * 128 + lane * 4);
        float* o = out + (long)d * 128 + lane * 4;
        atomicAdd(o + 0, v.x); atomicAdd(o + 1, v.y);
        atomicAdd(o + 2, v.z); atomicAdd(o + 3, v.w);
    }
}

__global__ void k_degree(const int* __restrict__ ed, float* __restrict__ deg, int E) {
    int i = blockIdx.x * blockDim.x + threadIdx.x;
    int st = gridDim.x * blockDim.x;
    for (; i < E; i += st) atomicAdd(&deg[ed[i]], 1.f);
}

// dst[n] = src[idx[n]] (row gather, wave per row)
__global__ void k_gather_rows(const float* __restrict__ src, const int* __restrict__ idx,
                              float* __restrict__ dst, int N) {
    int wid  = (blockIdx.x * blockDim.x + threadIdx.x) >> 5;
    int lane = threadIdx.x & 31;
    int nw   = (gridDim.x * blockDim.x) >> 5;
    for (int n = wid; n < N; n += nw) {
        int s = idx[n];
        *(float4*)(dst + (long)n * 128 + lane * 4) =
            *(const float4*)(src + (long)s * 128 + lane * 4);
    }
}

// root = deg>0 ? rsum/max(deg,1) : emb
__global__ void k_root(const float* __restrict__ rsum, const float* __restrict__ emb,
                       const float* __restrict__ deg, float* __restrict__ root, int N) {
    int wid  = (blockIdx.x * blockDim.x + threadIdx.x) >> 5;
    int lane = threadIdx.x & 31;
    int nw   = (gridDim.x * blockDim.x) >> 5;
    for (int n = wid; n < N; n += nw) {
        float d = deg[n];
        float inv = 1.f / fmaxf(d, 1.f);
        float4 r = *(const float4*)(rsum + (long)n * 128 + lane * 4);
        float4 e = *(const float4*)(emb + (long)n * 128 + lane * 4);
        float4 o;
        if (d > 0.f) { o.x = r.x * inv; o.y = r.y * inv; o.z = r.z * inv; o.w = r.w * inv; }
        else         { o = e; }
        *(float4*)(root + (long)n * 128 + lane * 4) = o;
    }
}

// ---------------- batch-norm (training mode, over node dim) ----------------
__global__ void k_bn_stats(const float* __restrict__ X, float* __restrict__ sum,
                           float* __restrict__ sumsq, int N) {
    int t = blockIdx.x * blockDim.x + threadIdx.x;
    int f = t & 127;
    int row = t >> 7;
    int rst = (gridDim.x * blockDim.x) >> 7;
    float s = 0.f, q = 0.f;
    for (int r = row; r < N; r += rst) {
        float x = X[(long)r * 128 + f];
        s += x; q += x * x;
    }
    atomicAdd(&sum[f], s); atomicAdd(&sumsq[f], q);
}

__global__ void k_bn_finalize(const float* __restrict__ sum, const float* __restrict__ sumsq,
                              const float* __restrict__ g, const float* __restrict__ b,
                              float* __restrict__ scale, float* __restrict__ shift, int N) {
    int f = threadIdx.x;
    float m = sum[f] / (float)N;
    float v = sumsq[f] / (float)N - m * m;
    float sc = g[f] * rsqrtf(v + 1e-5f);
    scale[f] = sc;
    shift[f] = b[f] - m * sc;
}

// y = relu(x*scale+shift), optionally accumulating stats of y for the next BN
__global__ void k_bn_apply(const float* __restrict__ X, const float* __restrict__ scale,
                           const float* __restrict__ shift, float* __restrict__ Y,
                           float* __restrict__ nsum, float* __restrict__ nsq,
                           int N, int collect) {
    int t = blockIdx.x * blockDim.x + threadIdx.x;
    int f = t & 127;
    int row = t >> 7;
    int rst = (gridDim.x * blockDim.x) >> 7;
    float sc = scale[f], sh = shift[f];
    float s = 0.f, q = 0.f;
    for (int r = row; r < N; r += rst) {
        float x = fmaxf(X[(long)r * 128 + f] * sc + sh, 0.f);
        Y[(long)r * 128 + f] = x;
        s += x; q += x * x;
    }
    if (collect) { atomicAdd(&nsum[f], s); atomicAdd(&nsq[f], q); }
}

// ---------------- node GEMM: Y = (alpha*X + AGG) @ W[128x128] + bias ----------------
// bf16 WMMA 16x16x32, f32 accumulate; B pre-swizzled into LDS fragments.
__global__ __launch_bounds__(256)
void k_gemm_node(const float* __restrict__ X, const float* __restrict__ AGG,
                 const float* __restrict__ epsP,
                 const float* __restrict__ W, const float* __restrict__ bias,
                 float* __restrict__ Y, int N) {
    __shared__ unsigned lb[4 * 8 * 32 * 8];  // 32 KB: [kt][nt][lane][8 dwords]
    int tid = threadIdx.x;
    for (int idx = tid; idx < 4 * 8 * 32 * 8; idx += blockDim.x) {
        int j = idx & 7; int lane = (idx >> 3) & 31; int nt = (idx >> 8) & 7; int kt = idx >> 11;
        int koff = (lane & 16) ? 8 : 0;
        int k = 32 * kt + ((j >> 2) << 4) + koff + ((j & 3) << 1);
        int n = 16 * nt + (lane & 15);
        lb[idx] = pkbf(W[k * 128 + n], W[(k + 1) * 128 + n]);
    }
    __syncthreads();
    float alpha = 1.0f + epsP[0];
    int lane = tid & 31;
    int m = lane & 15;
    int koff = (lane & 16) ? 8 : 0;
    int rowoff = (lane & 16) ? 8 : 0;  // output row offset within tile for acc rows
    // tile-invariant bias values for this lane's 8 output columns
    float bv[8];
#pragma unroll
    for (int nt = 0; nt < 8; nt++) bv[nt] = bias[nt * 16 + m];
    int wid = (blockIdx.x * blockDim.x + tid) >> 5;
    int nw  = (gridDim.x * blockDim.x) >> 5;
    int tiles = N >> 4;
    for (int t = wid; t < tiles; t += nw) {
        long row = (long)(t * 16 + m) * 128;
        v16bf A[4];
#pragma unroll
        for (int c = 0; c < 4; c++) {
            int b0 = 32 * c + koff;
            v16bf a;
#pragma unroll
            for (int i = 0; i < 8; i++)
                a[i] = (__bf16)(alpha * X[row + b0 + i] + AGG[row + b0 + i]);
#pragma unroll
            for (int i = 0; i < 8; i++)
                a[8 + i] = (__bf16)(alpha * X[row + b0 + 16 + i] + AGG[row + b0 + 16 + i]);
            A[c] = a;
        }
        // base pointer for this lane's output column; rows at 512B immediate offsets
        float* yb = Y + (long)(t * 16 + rowoff) * 128 + m;
#pragma unroll
        for (int nt = 0; nt < 8; nt++) {
            v8f acc = {0.f, 0.f, 0.f, 0.f, 0.f, 0.f, 0.f, 0.f};
#pragma unroll
            for (int kt = 0; kt < 4; kt++) {
                v8u raw = *(const v8u*)&lb[((kt * 8 + nt) * 32 + lane) * 8];
                v16bf B = __builtin_bit_cast(v16bf, raw);
                acc = __builtin_amdgcn_wmma_f32_16x16x32_bf16(
                    false, A[kt], false, B, (short)0, acc, false, false);
            }
            float* yp = yb + nt * 16;
            float bb = bv[nt];
#pragma unroll
            for (int r = 0; r < 8; r++) yp[r * 128] = acc[r] + bb;
        }
    }
}

// ---------------- edge scorer ----------------
// per edge: he=[root[src]‖feat[dst]] (1x256); z=relu(he@Wlin+blin); s=z·Wu+bu
// accumulates JSD term (pos or neg) into *accum. One wave per 16-edge tile.
__global__ __launch_bounds__(256)
void k_edge_score(const float* __restrict__ root, const float* __restrict__ feat,
                  const int* __restrict__ es, const int* __restrict__ ed,
                  const float* __restrict__ Wlin, const float* __restrict__ blin,
                  const float* __restrict__ Wu, const float* __restrict__ bu,
                  float* __restrict__ accum, int E, int neg) {
    __shared__ unsigned lb[8 * 8 * 32 * 8];  // 64 KB: [kt][nt][lane][8 dwords]
    int tid = threadIdx.x;
    for (int idx = tid; idx < 8 * 8 * 32 * 8; idx += blockDim.x) {
        int j = idx & 7; int lane = (idx >> 3) & 31; int nt = (idx >> 8) & 7; int kt = idx >> 11;
        int koff = (lane & 16) ? 8 : 0;
        int k = 32 * kt + ((j >> 2) << 4) + koff + ((j & 3) << 1);
        int n = 16 * nt + (lane & 15);
        lb[idx] = pkbf(Wlin[k * 128 + n], Wlin[(k + 1) * 128 + n]);
    }
    __syncthreads();
    int lane = tid & 31;
    int m = lane & 15;
    int koff = (lane & 16) ? 8 : 0;
    // tile-invariant epilogue constants for this lane's 8 output columns
    float blv[8], wuv[8];
#pragma unroll
    for (int nt = 0; nt < 8; nt++) { blv[nt] = blin[nt * 16 + m]; wuv[nt] = Wu[nt * 16 + m]; }
    float bu0 = bu[0];
    int wid = (blockIdx.x * blockDim.x + tid) >> 5;
    int nw  = (gridDim.x * blockDim.x) >> 5;
    int tiles = E >> 4;
    const float LOG2C = 0.6931471805599453f;
    float wacc = 0.f;
    for (int t = wid; t < tiles; t += nw) {
        int e = t * 16 + m;
        long srow = (long)es[e] * 128;
        long drow = (long)ed[e] * 128;
        v16bf A[8];
#pragma unroll
        for (int c = 0; c < 4; c++) {   // K 0..127: root[src]
            int b0 = 32 * c + koff;
            v16bf a;
#pragma unroll
            for (int i = 0; i < 8; i++) a[i]     = (__bf16)root[srow + b0 + i];
#pragma unroll
            for (int i = 0; i < 8; i++) a[8 + i] = (__bf16)root[srow + b0 + 16 + i];
            A[c] = a;
        }
#pragma unroll
        for (int c = 0; c < 4; c++) {   // K 128..255: feat[dst]
            int b0 = 32 * c + koff;
            v16bf a;
#pragma unroll
            for (int i = 0; i < 8; i++) a[i]     = (__bf16)feat[drow + b0 + i];
#pragma unroll
            for (int i = 0; i < 8; i++) a[8 + i] = (__bf16)feat[drow + b0 + 16 + i];
            A[4 + c] = a;
        }
        float pvt[8];
#pragma unroll
        for (int r = 0; r < 8; r++) pvt[r] = 0.f;
#pragma unroll
        for (int nt = 0; nt < 8; nt++) {
            v8f acc = {0.f, 0.f, 0.f, 0.f, 0.f, 0.f, 0.f, 0.f};
#pragma unroll
            for (int kt = 0; kt < 8; kt++) {
                v8u raw = *(const v8u*)&lb[((kt * 8 + nt) * 32 + lane) * 8];
                v16bf B = __builtin_bit_cast(v16bf, raw);
                acc = __builtin_amdgcn_wmma_f32_16x16x32_bf16(
                    false, A[kt], false, B, (short)0, acc, false, false);
            }
            float bl = blv[nt];
            float wu = wuv[nt];
#pragma unroll
            for (int r = 0; r < 8; r++) {
                float z = fmaxf(acc[r] + bl, 0.f);
                pvt[r] += z * wu;
            }
        }
        // reduce over the 16 lanes of each half (cols of the tile)
#pragma unroll
        for (int r = 0; r < 8; r++) {
            pvt[r] += __shfl_xor(pvt[r], 1);
            pvt[r] += __shfl_xor(pvt[r], 2);
            pvt[r] += __shfl_xor(pvt[r], 4);
            pvt[r] += __shfl_xor(pvt[r], 8);
        }
        // lanes with m<8 finalize one edge each: rows 0..7 (lo half) / 8..15 (hi half)
        float s = 0.f;
#pragma unroll
        for (int r = 0; r < 8; r++) if (m == r) s = pvt[r];
        s += bu0;
        float f = softplus_f(-s);
        float term = neg ? (f + s - LOG2C) : (LOG2C - f);
        if (m >= 8) term = 0.f;
        wacc += term;
    }
    wacc += __shfl_xor(wacc, 1);
    wacc += __shfl_xor(wacc, 2);
    wacc += __shfl_xor(wacc, 4);
    wacc += __shfl_xor(wacc, 8);
    wacc += __shfl_xor(wacc, 16);
    if (lane == 0) atomicAdd(accum, wacc);
}

__global__ void k_final(const float* __restrict__ accum, float* __restrict__ out, float invE) {
    out[0] = (accum[1] - accum[0]) * invE;
}

// ---------------- launch ----------------
extern "C" void kernel_launch(void* const* d_in, const int* in_sizes, int n_in,
                              void* d_out, int out_size, void* d_ws, size_t ws_size,
                              hipStream_t stream) {
    const float* features = (const float*)d_in[0];
    const float* gin_W    = (const float*)d_in[1];
    const float* gin_b    = (const float*)d_in[2];
    const float* gin_eps  = (const float*)d_in[3];
    const float* bn1_g    = (const float*)d_in[4];
    const float* bn1_b    = (const float*)d_in[5];
    const float* bn2_g    = (const float*)d_in[6];
    const float* bn2_b    = (const float*)d_in[7];
    // d_in[8]=Wz, d_in[9]=bz: dead code in the reference (m1 deleted) -> skipped
    const float* Wlin     = (const float*)d_in[10];
    const float* blin     = (const float*)d_in[11];
    const float* Wu       = (const float*)d_in[12];
    const float* bu       = (const float*)d_in[13];
    const int*   es       = (const int*)d_in[14];
    const int*   ed       = (const int*)d_in[15];
    const int*   perm     = (const int*)d_in[16];

    int N = in_sizes[0] / 128;
    int E = in_sizes[14];
    size_t nf = (size_t)N * 128;

    float* ws    = (float*)d_ws;
    float* buf0  = ws;            // node buffer A
    float* buf1  = buf0 + nf;     // node buffer B / hperm
    float* buf2  = buf1 + nf;     // agg / rsum
    float* buf3  = buf2 + nf;     // root
    float* deg   = buf3 + nf;     // [N]
    float* s1    = deg + N;       // 4x128 stats (contiguous)
    float* q1    = s1 + 128;
    float* s2    = q1 + 128;
    float* q2    = s2 + 128;
    float* scale = q2 + 128;
    float* shift = scale + 128;
    float* accum = shift + 128;   // [2] pos,neg

    const float* h = features;
    float* ybuf[3] = {buf0, buf1, buf0};

    for (int l = 0; l < 3; l++) {
        k_zero<<<2048, 256, 0, stream>>>(buf2, (long)nf);
        k_scatter_add<<<8192, 256, 0, stream>>>(h, es, ed, buf2, E);
        float* y = ybuf[l];
        k_gemm_node<<<392, 256, 0, stream>>>(h, buf2, gin_eps + l, gin_W + (size_t)l * 128 * 128,
                                             gin_b + l * 128, y, N);
        k_zero<<<1, 256, 0, stream>>>(s1, 512);  // zeros s1,q1,s2,q2
        k_bn_stats<<<256, 256, 0, stream>>>(y, s1, q1, N);
        k_bn_finalize<<<1, 128, 0, stream>>>(s1, q1, bn1_g + l * 128, bn1_b + l * 128, scale, shift, N);
        k_bn_apply<<<256, 256, 0, stream>>>(y, scale, shift, y, s2, q2, N, 1);
        k_bn_finalize<<<1, 128, 0, stream>>>(s2, q2, bn2_g + l * 128, bn2_b + l * 128, scale, shift, N);
        k_bn_apply<<<256, 256, 0, stream>>>(y, scale, shift, y, nullptr, nullptr, N, 0);
        h = y;
    }
    // h == buf0 (final embedding)

    k_zero<<<256, 256, 0, stream>>>(deg, (long)N);
    k_degree<<<2048, 256, 0, stream>>>(ed, deg, E);
    k_zero<<<1, 64, 0, stream>>>(accum, 2);

    // positive
    k_zero<<<2048, 256, 0, stream>>>(buf2, (long)nf);
    k_scatter_add<<<8192, 256, 0, stream>>>(h, es, ed, buf2, E);
    k_root<<<2048, 256, 0, stream>>>(buf2, h, deg, buf3, N);
    k_edge_score<<<1024, 256, 0, stream>>>(buf3, features, es, ed, Wlin, blin, Wu, bu,
                                           accum + 0, E, 0);
    // negative: hperm = h[perm]
    k_gather_rows<<<2048, 256, 0, stream>>>(h, perm, buf1, N);
    k_zero<<<2048, 256, 0, stream>>>(buf2, (long)nf);
    k_scatter_add<<<8192, 256, 0, stream>>>(buf1, es, ed, buf2, E);
    k_root<<<2048, 256, 0, stream>>>(buf2, buf1, deg, buf3, N);
    k_edge_score<<<1024, 256, 0, stream>>>(buf3, features, es, ed, Wlin, blin, Wu, bu,
                                           accum + 1, E, 1);

    k_final<<<1, 1, 0, stream>>>(accum, (float*)d_out, 1.0f / (float)E);
}
